// DeformMambaEncoderLayer_88029649699169
// MI455X (gfx1250) — matrix operand order repaired and checked by reading, not compiled, not run on hardware
//
#include <hip/hip_runtime.h>
#include <hip/hip_bf16.h>
#include <math.h>

// ---------------- problem constants (match reference) ----------------
#define B_   4
#define C_   128
#define H_   64
#define W_   64
#define L_   4096          // H_*W_
#define F_   512
#define DI_  256
#define N_   16
#define R_   8
#define K_   4
#define G_   2
#define KK_  9

typedef __attribute__((ext_vector_type(16))) __bf16 v16bf;
typedef __attribute__((ext_vector_type(8)))  float  v8f;

union FragU {
    unsigned int u[8];
    v16bf v;
};

__device__ __forceinline__ unsigned int f2bf1(float f) {
    unsigned int u = __float_as_uint(f);
    u += 0x7FFFu + ((u >> 16) & 1u);      // round-to-nearest-even
    return u >> 16;
}

// async global->LDS copy of 16 bytes (gfx1250 GLOBAL_LOAD_ASYNC_TO_LDS_B128)
__device__ __forceinline__ void async_ldsb128(unsigned lds_addr,
                                              unsigned long long gaddr) {
    asm volatile("global_load_async_to_lds_b128 %0, %1, off"
                 :: "v"(lds_addr), "v"(gaddr) : "memory");
}
__device__ __forceinline__ void wait_async0() {
    asm volatile("s_wait_asynccnt 0" ::: "memory");
}

// ---------------------------------------------------------------------
// Generic (optionally batched) GEMM:  C[m,n] = sum_k A[m,k] * W[n,k] + bias[n]
// A: f32 activations (M multiple of 64), converted to bf16 into padded LDS.
// W: pre-converted bf16 weights (N x K row-major), staged to LDS by the
//    async tensor-data path (global_load_async_to_lds_b128, ASYNCcnt).
// 128 threads = 4 waves; 64x64 tile; K-step 32; 4 x v_wmma per step.
// ---------------------------------------------------------------------
#define LDSS 36   // u16 stride for A tile: 72B rows -> conflict-free u32 reads

__global__ __launch_bounds__(128) void k_gemm_bf16(
    const float* __restrict__ A, int lda, long aStride,
    const unsigned short* __restrict__ Wm, int ldw, long wStride, int wMod,
    const float* __restrict__ bias,
    float* __restrict__ Cd, int ldc, long cStride,
    int M, int N, int Kdim)
{
    __shared__ unsigned short As[64 * LDSS];
    __shared__ __align__(16) unsigned short Bs[64 * 32];

    const float*          Ab = A  + (size_t)blockIdx.z * (size_t)aStride;
    const unsigned short* Wb = Wm + (size_t)(blockIdx.z % wMod) * (size_t)wStride;
    float*                Cb = Cd + (size_t)blockIdx.z * (size_t)cStride;

    const int m0  = blockIdx.y * 64;
    const int n0  = blockIdx.x * 64;
    const int tid = threadIdx.x;
    const int lane = tid & 31;
    const int wv   = tid >> 5;
    const int half = lane >> 4;
    const int l16  = lane & 15;

    const unsigned bsBase = (unsigned)(size_t)(void*)&Bs[0];

    v8f acc[4];
#pragma unroll
    for (int j = 0; j < 4; j++)
#pragma unroll
        for (int e = 0; e < 8; e++) acc[j][e] = 0.0f;

    for (int k0 = 0; k0 < Kdim; k0 += 32) {
        // --- async-stage W tile: 64 rows x 64B, 2 x B128 per thread ---
#pragma unroll
        for (int i = tid; i < 256; i += 128) {
            int r  = i >> 2;
            int ch = i & 3;
            int gn = n0 + r;
            if (gn >= N) gn = N - 1;               // clamp: stay in-bounds
            unsigned long long ga =
                (unsigned long long)(size_t)(Wb + (size_t)gn * ldw + k0) +
                (unsigned long long)(ch * 16);
            async_ldsb128(bsBase + (unsigned)(r * 64 + ch * 16), ga);
        }
        // prefetch next A K-tile (global_prefetch_b8)
        if (k0 + 32 < Kdim) {
            __builtin_prefetch(Ab + (size_t)(m0 + (tid >> 1)) * lda +
                               (k0 + 32) + (tid & 1) * 16, 0, 0);
        }
        // --- stage A tile (64x32 f32 -> bf16), vectorized pack ---
        for (int i = tid; i < 64 * 16; i += 128) {
            int r  = i >> 4;
            int cp = i & 15;
            const float2 f2 =
                *(const float2*)(Ab + (size_t)(m0 + r) * lda + k0 + 2 * cp);
            unsigned pk = f2bf1(f2.x) | (f2bf1(f2.y) << 16);
            ((unsigned*)As)[r * (LDSS / 2) + cp] = pk;
        }
        wait_async0();
        __syncthreads();

        // A fragment: lane half h holds K = h*8..h*8+7 (v0-3) and 16+h*8.. (v4-7)
        FragU a;
        const unsigned int* Ap = (const unsigned int*)(As + (wv * 16 + l16) * LDSS);
        const int kb = half * 4;
#pragma unroll
        for (int r = 0; r < 4; r++) {
            a.u[r]     = Ap[kb + r];
            a.u[4 + r] = Ap[8 + kb + r];
        }
#pragma unroll
        for (int j = 0; j < 4; j++) {
            // B fragment: lane half h holds K = h*8..h*8+7 (v0-3) and 16+h*8.. (v4-7)
            FragU bf;
            const unsigned int* Bp = (const unsigned int*)(Bs + (j * 16 + l16) * 32);
#pragma unroll
            for (int r = 0; r < 8; r++) bf.u[r] = Bp[(r >> 2) * 8 + half * 4 + (r & 3)];
            acc[j] = __builtin_amdgcn_wmma_f32_16x16x32_bf16(
                false, a.v, false, bf.v, (short)0, acc[j], false, false);
        }
        __syncthreads();
    }

    // C/D layout: VGPR r, lane half h -> row r + 8*h ; col = lane&15
#pragma unroll
    for (int j = 0; j < 4; j++) {
        int n = n0 + j * 16 + l16;
        if (n < N) {
            float bv = bias ? bias[n] : 0.0f;
#pragma unroll
            for (int r = 0; r < 8; r++) {
                int m = m0 + wv * 16 + half * 8 + r;
                Cb[(size_t)m * ldc + n] = acc[j][r] + bv;
            }
        }
    }
}

// ---------------------------------------------------------------------
// f32 -> bf16 weight conversion (one-shot, tiny)
// ---------------------------------------------------------------------
__global__ void k_f2bf(const float* __restrict__ in,
                       unsigned short* __restrict__ out, int n)
{
    int t = blockIdx.x * blockDim.x + threadIdx.x;
    if (t < n) out[t] = (unsigned short)f2bf1(in[t]);
}

// ---------------------------------------------------------------------
// LayerNorm over last dim (32-thread wave per row), optional residual add.
// ---------------------------------------------------------------------
__global__ __launch_bounds__(32) void k_layernorm(
    const float* __restrict__ in, const float* __restrict__ res,
    const float* __restrict__ g, const float* __restrict__ bt,
    float* __restrict__ out, int Cw)
{
    const int row  = blockIdx.x;
    const int lane = threadIdx.x;
    const float* p = in + (size_t)row * Cw;
    float vals[8];
    const int nper = Cw >> 5;
    float s = 0.f, s2 = 0.f;
    for (int i = 0; i < nper; i++) {
        float v = p[lane + 32 * i];
        vals[i] = v; s += v; s2 += v * v;
    }
    for (int off = 16; off; off >>= 1) {
        s  += __shfl_xor(s,  off, 32);
        s2 += __shfl_xor(s2, off, 32);
    }
    float mu   = s / (float)Cw;
    float var  = s2 / (float)Cw - mu * mu;
    float rstd = rsqrtf(var + 1e-5f);
    float* q = out + (size_t)row * Cw;
    const float* rr = res ? res + (size_t)row * Cw : nullptr;
    for (int i = 0; i < nper; i++) {
        int c = lane + 32 * i;
        float v = (vals[i] - mu) * rstd * g[c] + bt[c];
        q[c] = rr ? rr[c] + v : v;
    }
}

// ---------------------------------------------------------------------
// Depthwise 3x3 conv (SAME, zero pad) + bias + activation.
// Layout (B, L, Ch) channel-innermost; ld = row stride (strided views ok).
// act: 1 gelu(exact), 2 silu
// ---------------------------------------------------------------------
__global__ void k_dwconv_act(
    const float* __restrict__ in, int ldin,
    const float* __restrict__ wgt, const float* __restrict__ bias,
    float* __restrict__ out, int ldout, int Ch, int act, int total)
{
    int t = blockIdx.x * blockDim.x + threadIdx.x;
    if (t >= total) return;
    int c = t % Ch;
    int l = (t / Ch) % L_;
    int b = t / (Ch * L_);
    int h = l >> 6, w = l & 63;
    const float* ib = in + (size_t)b * L_ * ldin + c;
    float s = 0.f;
#pragma unroll
    for (int dy = -1; dy <= 1; dy++)
#pragma unroll
        for (int dx = -1; dx <= 1; dx++) {
            int hh = h + dy, ww = w + dx;
            if (hh >= 0 && hh < H_ && ww >= 0 && ww < W_)
                s += wgt[c * 9 + (dy + 1) * 3 + (dx + 1)] *
                     ib[(size_t)(hh * W_ + ww) * ldin];
        }
    s += bias[c];
    if (act == 1)      s = 0.5f * s * (1.f + erff(s * 0.70710678118654752f));
    else if (act == 2) s = s / (1.f + expf(-s));
    out[((size_t)b * L_ + l) * ldout + c] = s;
}

// ---------------------------------------------------------------------
// DCNv4 bilinear deformable sampling. thread per (b,l,g,c), Cg=64.
// ---------------------------------------------------------------------
__global__ void k_dcn_sample(const float* __restrict__ v,
                             const float* __restrict__ om,
                             float* __restrict__ samp)
{
    int t = blockIdx.x * blockDim.x + threadIdx.x;
    if (t >= B_ * L_ * C_) return;
    int c = t & 63;
    int g = (t >> 6) & 1;
    int l = (t >> 7) & (L_ - 1);
    int b = t >> 19;
    int h = l >> 6, w = l & 63;
    const float* omr = om + ((size_t)b * L_ + l) * (G_ * 3 * KK_) + g * (3 * KK_);
    const float* vb  = v + (size_t)b * L_ * C_ + g * 64 + c;
    float acc = 0.f;
#pragma unroll
    for (int kk = 0; kk < KK_; kk++) {
        float offy = omr[kk * 2 + 0];
        float offx = omr[kk * 2 + 1];
        float mask = omr[2 * KK_ + kk];
        float py = (float)(h + kk / 3 - 1) + offy;
        float px = (float)(w + kk % 3 - 1) + offx;
        float y0 = floorf(py), x0 = floorf(px);
        float wy = py - y0, wx = px - x0;
        float s = 0.f;
#pragma unroll
        for (int dy = 0; dy < 2; dy++)
#pragma unroll
            for (int dx = 0; dx < 2; dx++) {
                float yy = y0 + (float)dy, xx = x0 + (float)dx;
                float valid = (yy >= 0.f && yy < (float)H_ &&
                               xx >= 0.f && xx < (float)W_) ? 1.f : 0.f;
                float wg = (dy ? wy : 1.f - wy) * (dx ? wx : 1.f - wx) * valid;
                int yc = (int)fminf(fmaxf(yy, 0.f), (float)(H_ - 1));
                int xc = (int)fminf(fmaxf(xx, 0.f), (float)(W_ - 1));
                s += wg * vb[(size_t)(yc * W_ + xc) * C_];
            }
        acc += mask * s;
    }
    samp[((size_t)b * L_ + l) * C_ + g * 64 + c] = acc;
}

// ---------------------------------------------------------------------
// Build 4-direction sequences xs (B,K,L,DI) from xc (B,L,DI) by scatter.
// ---------------------------------------------------------------------
__global__ void k_build_xs(const float* __restrict__ xc, float* __restrict__ xs)
{
    int t = blockIdx.x * blockDim.x + threadIdx.x;
    if (t >= B_ * L_ * DI_) return;
    int d = t & (DI_ - 1);
    int l = (t >> 8) & (L_ - 1);
    int b = t >> 20;
    float v = xc[((size_t)b * L_ + l) * DI_ + d];
    int h = l >> 6, w = l & 63;
    int l1 = (w << 6) | h;
    size_t base = (size_t)b * K_ * L_ * DI_;
    xs[base + ((size_t)0 * L_ + l)             * DI_ + d] = v;
    xs[base + ((size_t)1 * L_ + l1)            * DI_ + d] = v;
    xs[base + ((size_t)2 * L_ + (L_ - 1 - l))  * DI_ + d] = v;
    xs[base + ((size_t)3 * L_ + (L_ - 1 - l1)) * DI_ + d] = v;
}

// ---------------------------------------------------------------------
// Selective scan: one block per (b,k); 256 threads = DI channels.
// dbl row = [dts(8) | B(16) | C(16)] staged into LDS per step.
// delta computed in-register: softplus(dts . dt_w[k,d,:] + dt_b[k,d]).
// ---------------------------------------------------------------------
__global__ __launch_bounds__(256) void k_scan(
    const float* __restrict__ dbl, const float* __restrict__ xs,
    const float* __restrict__ dt_w, const float* __restrict__ dt_b,
    const float* __restrict__ A_logs, const float* __restrict__ Ds,
    float* __restrict__ ys)
{
    const int bk = blockIdx.x;          // b*K + k
    const int k  = bk & (K_ - 1);
    const int d  = threadIdx.x;
    const int kd = k * DI_ + d;

    float dtw[R_];
#pragma unroll
    for (int r = 0; r < R_; r++) dtw[r] = dt_w[(size_t)kd * R_ + r];
    const float dtb = dt_b[kd];
    float Av[N_];
#pragma unroll
    for (int n = 0; n < N_; n++) Av[n] = -expf(A_logs[(size_t)kd * N_ + n]);
    const float Dv = Ds[kd];

    float hs[N_];
#pragma unroll
    for (int n = 0; n < N_; n++) hs[n] = 0.f;

    __shared__ float sh[40];
    const size_t rowbase = (size_t)bk * L_;

    for (int l = 0; l < L_; l++) {
        const float* dr = dbl + (rowbase + l) * 40;
        if (d < 40) sh[d] = dr[d];
        __syncthreads();
        float u = xs[(rowbase + l) * DI_ + d];
        float x = dtb;
#pragma unroll
        for (int r = 0; r < R_; r++) x += sh[r] * dtw[r];
        float delta = (x > 20.f) ? x : log1pf(expf(x));
        float du = delta * u;
        float y = 0.f;
#pragma unroll
        for (int n = 0; n < N_; n++) {
            hs[n] = __expf(delta * Av[n]) * hs[n] + du * sh[8 + n];
            y += hs[n] * sh[24 + n];
        }
        ys[(rowbase + l) * DI_ + d] = y + u * Dv;
        __syncthreads();
    }
}

// ---------------------------------------------------------------------
// Merge 4 directions + layernorm(onorm) + silu(z) gate.  32 threads/row.
// ---------------------------------------------------------------------
__global__ __launch_bounds__(32) void k_merge_gate(
    const float* __restrict__ ys, const float* __restrict__ xz,
    const float* __restrict__ ng, const float* __restrict__ nb,
    float* __restrict__ yg)
{
    const int row = blockIdx.x;           // b*L + l
    const int b = row >> 12;
    const int l = row & (L_ - 1);
    const int lane = threadIdx.x;
    const int h = l >> 6, w = l & 63;
    const int l1 = (w << 6) | h;
    const size_t base = (size_t)b * K_ * L_ * DI_;
    const float* r0 = ys + base + (size_t)l * DI_;
    const float* r1 = ys + base + ((size_t)L_ + l1) * DI_;
    const float* r2 = ys + base + ((size_t)2 * L_ + (L_ - 1 - l)) * DI_;
    const float* r3 = ys + base + ((size_t)3 * L_ + (L_ - 1 - l1)) * DI_;

    float vals[8];
    float s = 0.f, s2 = 0.f;
#pragma unroll
    for (int i = 0; i < 8; i++) {
        int d = lane + 32 * i;
        float v = r0[d] + r1[d] + r2[d] + r3[d];
        vals[i] = v; s += v; s2 += v * v;
    }
    for (int off = 16; off; off >>= 1) {
        s  += __shfl_xor(s,  off, 32);
        s2 += __shfl_xor(s2, off, 32);
    }
    float mu   = s / (float)DI_;
    float var  = s2 / (float)DI_ - mu * mu;
    float rstd = rsqrtf(var + 1e-5f);
    const float* z = xz + (size_t)row * (2 * DI_) + DI_;
    float* q = yg + (size_t)row * DI_;
#pragma unroll
    for (int i = 0; i < 8; i++) {
        int d = lane + 32 * i;
        float zn = z[d];
        q[d] = ((vals[i] - mu) * rstd * ng[d] + nb[d]) * (zn / (1.f + expf(-zn)));
    }
}

__global__ void k_final_add(const float* __restrict__ a,
                            const float* __restrict__ b2,
                            const float* __restrict__ c2,
                            float* __restrict__ out, int n)
{
    int t = blockIdx.x * blockDim.x + threadIdx.x;
    if (t < n) out[t] = a[t] + b2[t] + c2[t];
}

// ---------------------------------------------------------------------
// Launcher
// ---------------------------------------------------------------------
static inline void gemm_launch(hipStream_t s,
    const float* A, int lda, long aStride,
    const unsigned short* Wm, int ldw, long wStride, int wMod,
    const float* bias, float* Cd, int ldc, long cStride,
    int M, int N, int Kdim, int batches)
{
    dim3 grid((N + 63) / 64, (M + 63) / 64, batches);
    k_gemm_bf16<<<grid, 128, 0, s>>>(A, lda, aStride, Wm, ldw, wStride, wMod,
                                     bias, Cd, ldc, cStride, M, N, Kdim);
}

static inline void conv_w(hipStream_t s, const float* in, unsigned short* out, int n)
{
    k_f2bf<<<(n + 255) / 256, 256, 0, s>>>(in, out, n);
}

extern "C" void kernel_launch(void* const* d_in, const int* in_sizes, int n_in,
                              void* d_out, int out_size, void* d_ws, size_t ws_size,
                              hipStream_t stream)
{
    (void)in_sizes; (void)n_in; (void)out_size; (void)ws_size;
    const float* x        = (const float*)d_in[0];
    const float* ln1_g    = (const float*)d_in[3];
    const float* ln1_b    = (const float*)d_in[4];
    const float* dcn_vw   = (const float*)d_in[5];
    const float* dcn_vb   = (const float*)d_in[6];
    const float* dcn_omw  = (const float*)d_in[7];
    const float* dcn_omb  = (const float*)d_in[8];
    const float* dcn_ow   = (const float*)d_in[9];
    const float* dcn_ob   = (const float*)d_in[10];
    const float* dn1_g    = (const float*)d_in[11];
    const float* dn1_b    = (const float*)d_in[12];
    const float* dn2_g    = (const float*)d_in[13];
    const float* dn2_b    = (const float*)d_in[14];
    const float* dmf_fc1w = (const float*)d_in[15];
    const float* dmf_fc1b = (const float*)d_in[16];
    const float* dmf_pew  = (const float*)d_in[17];
    const float* dmf_peb  = (const float*)d_in[18];
    const float* dmf_fc2w = (const float*)d_in[19];
    const float* dmf_fc2b = (const float*)d_in[20];
    const float* in_w     = (const float*)d_in[21];
    const float* conv_wv  = (const float*)d_in[22];
    const float* conv_b   = (const float*)d_in[23];
    const float* xproj_w  = (const float*)d_in[24];
    const float* dt_w     = (const float*)d_in[25];
    const float* dt_b     = (const float*)d_in[26];
    const float* A_logs   = (const float*)d_in[27];
    const float* Ds       = (const float*)d_in[28];
    const float* onorm_g  = (const float*)d_in[29];
    const float* onorm_b  = (const float*)d_in[30];
    const float* out_w    = (const float*)d_in[31];
    const float* ln2_g    = (const float*)d_in[32];
    const float* ln2_b    = (const float*)d_in[33];
    const float* mf_fc1w  = (const float*)d_in[34];
    const float* mf_fc1b  = (const float*)d_in[35];
    const float* mf_pew   = (const float*)d_in[36];
    const float* mf_peb   = (const float*)d_in[37];
    const float* mf_fc2w  = (const float*)d_in[38];
    const float* mf_fc2b  = (const float*)d_in[39];

    float* ws = (float*)d_ws;
    const size_t BLC   = (size_t)B_ * L_ * C_;        // 2,097,152
    const size_t BL512 = (size_t)B_ * L_ * 512;       // 8,388,608
    const size_t BLDI  = (size_t)B_ * L_ * DI_;       // 4,194,304

    // workspace overlay (floats); peak ~254 MB
    float* xln  = ws + 0;                       // later reused as x4
    float* x1   = ws + 2097152;
    float* x2   = ws + 4194304;
    float* bufA = ws + 6291456;                 // h1 (ffn1), then xz (ossm)
    float* bufB = ws + 14680064;                // g1 (ffn1)
    float* dblb = bufB;                         // (B,K,L,40) after g1 dies
    float* xc   = bufB + 3145728;               // (B,L,DI)
    float* pool = ws + 23068672;                // v+samp, later yg
    float* vbuf = pool;
    float* samp = pool + BLC;
    float* yg   = pool;
    float* omb  = ws + 27262976;                // (B,L,54)
    float* dcnr = ws + 28147712;
    float* tmp  = ws + 30244864;                // f2 / x3 / f3
    float* xsb  = ws + 32342016;                // (B,K,L,DI); later h2
    float* ysb  = ws + 49119232;                // (B,K,L,DI); later g2
    float* x4   = xln;
    float* h2   = xsb;
    float* g2   = ysb;

    // bf16 weight pool (u16 elements), 16B-aligned region at ws tail
    unsigned short* wbf = (unsigned short*)(ws + 65896448);
    unsigned short* w_vw   = wbf + 0;         // 128*128
    unsigned short* w_omw  = wbf + 16384;     // 54*128
    unsigned short* w_ow   = wbf + 23296;     // 128*128
    unsigned short* w_dfc1 = wbf + 39680;     // 512*128
    unsigned short* w_dfc2 = wbf + 105216;    // 128*512
    unsigned short* w_inw  = wbf + 170752;    // 512*128
    unsigned short* w_xprj = wbf + 236288;    // 4*40*256
    unsigned short* w_outw = wbf + 277248;    // 128*256
    unsigned short* w_mfc1 = wbf + 310016;    // 512*128
    unsigned short* w_mfc2 = wbf + 375552;    // 128*512

    const int BL = B_ * L_;

    // ---- weight conversion (one-shot, tiny) ----
    conv_w(stream, dcn_vw,   w_vw,   C_ * C_);
    conv_w(stream, dcn_omw,  w_omw,  54 * C_);
    conv_w(stream, dcn_ow,   w_ow,   C_ * C_);
    conv_w(stream, dmf_fc1w, w_dfc1, F_ * C_);
    conv_w(stream, dmf_fc2w, w_dfc2, C_ * F_);
    conv_w(stream, in_w,     w_inw,  2 * DI_ * C_);
    conv_w(stream, xproj_w,  w_xprj, K_ * (R_ + 2 * N_) * DI_);
    conv_w(stream, out_w,    w_outw, C_ * DI_);
    conv_w(stream, mf_fc1w,  w_mfc1, F_ * C_);
    conv_w(stream, mf_fc2w,  w_mfc2, C_ * F_);

    // ---- phase 0: xln = LN(x) ----
    k_layernorm<<<BL, 32, 0, stream>>>(x, nullptr, ln1_g, ln1_b, xln, C_);

    // ---- phase 1: DCNv4 ----
    gemm_launch(stream, xln, C_, 0, w_vw, C_, 0, 1, dcn_vb, vbuf, C_, 0,
                BL, C_, C_, 1);
    gemm_launch(stream, xln, C_, 0, w_omw, C_, 0, 1, dcn_omb, omb, 54, 0,
                BL, 54, C_, 1);
    {
        int tot = B_ * L_ * C_;
        k_dcn_sample<<<(tot + 255) / 256, 256, 0, stream>>>(vbuf, omb, samp);
    }
    gemm_launch(stream, samp, C_, 0, w_ow, C_, 0, 1, dcn_ob, dcnr, C_, 0,
                BL, C_, C_, 1);
    k_layernorm<<<BL, 32, 0, stream>>>(dcnr, xln, dn1_g, dn1_b, x1, C_);

    // ---- phase 2: conv-FFN #1 ----
    gemm_launch(stream, x1, C_, 0, w_dfc1, C_, 0, 1, dmf_fc1b, bufA, F_, 0,
                BL, F_, C_, 1);
    k_dwconv_act<<<(int)((BL512 + 255) / 256), 256, 0, stream>>>(
        bufA, F_, dmf_pew, dmf_peb, bufB, F_, F_, 1, (int)BL512);
    gemm_launch(stream, bufB, F_, 0, w_dfc2, F_, 0, 1, dmf_fc2b, tmp, C_, 0,
                BL, C_, F_, 1);
    k_layernorm<<<BL, 32, 0, stream>>>(tmp, x1, dn2_g, dn2_b, x2, C_);

    // ---- phase 3: OSSM ----
    float* xz = bufA;
    gemm_launch(stream, x2, C_, 0, w_inw, C_, 0, 1, nullptr, xz, 2 * DI_, 0,
                BL, 2 * DI_, C_, 1);
    k_dwconv_act<<<(int)((BLDI + 255) / 256), 256, 0, stream>>>(
        xz, 2 * DI_, conv_wv, conv_b, xc, DI_, DI_, 2, (int)BLDI);
    k_build_xs<<<(int)((BLDI + 255) / 256), 256, 0, stream>>>(xc, xsb);
    // per-direction x-projection: 16 batches (b,k), weights modulo K
    gemm_launch(stream, xsb, DI_, (long)L_ * DI_,
                w_xprj, DI_, (long)(R_ + 2 * N_) * DI_, K_,
                nullptr, dblb, 40, (long)L_ * 40,
                L_, R_ + 2 * N_, DI_, B_ * K_);
    k_scan<<<B_ * K_, DI_, 0, stream>>>(dblb, xsb, dt_w, dt_b, A_logs, Ds, ysb);
    k_merge_gate<<<BL, 32, 0, stream>>>(ysb, xz, onorm_g, onorm_b, yg);
    gemm_launch(stream, yg, DI_, 0, w_outw, DI_, 0, 1, nullptr, tmp, C_, 0,
                BL, C_, DI_, 1);
    k_layernorm<<<BL, 32, 0, stream>>>(tmp, nullptr, ln2_g, ln2_b, x4, C_);

    // ---- phase 4: conv-FFN #2 ----
    gemm_launch(stream, x4, C_, 0, w_mfc1, C_, 0, 1, mf_fc1b, h2, F_, 0,
                BL, F_, C_, 1);
    k_dwconv_act<<<(int)((BL512 + 255) / 256), 256, 0, stream>>>(
        h2, F_, mf_pew, mf_peb, g2, F_, F_, 1, (int)BL512);
    gemm_launch(stream, g2, F_, 0, w_mfc2, F_, 0, 1, mf_fc2b, tmp, C_, 0,
                BL, C_, F_, 1);

    // ---- output: identity + x4 + ffn2 ----
    k_final_add<<<(int)((BLC + 255) / 256), 256, 0, stream>>>(
        x, x4, tmp, (float*)d_out, (int)BLC);
}
